// ResidualVectorQuantizer_21191368638523
// MI455X (gfx1250) — matrix operand order repaired and checked by reading, not compiled, not run on hardware
//
#include <hip/hip_runtime.h>
#include <hip/hip_bf16.h>

// Residual Vector Quantization (Encodec-style) for MI455X / gfx1250.
// Core: fused bf16-WMMA distance GEMM + argmin per layer, with the codebook
// tile staged into LDS via the CDNA5 async copy engine (double-buffered).

typedef __attribute__((ext_vector_type(16))) __bf16        v16bf;
typedef __attribute__((ext_vector_type(8)))  float         v8f;
typedef __attribute__((ext_vector_type(4)))  unsigned int  u32x4;

#define BQ 8        // n_q
#define NB 1024     // bins
#define DD 256      // D
#define BB 16       // B
#define TT 4096     // T
#define NN (BB*TT)  // 65536 vectors

__device__ __forceinline__ unsigned short f2bf(float f) {
  // round-to-nearest-even float -> bf16 (finite inputs)
  unsigned int u = __float_as_uint(f);
  unsigned int r = u + 0x7FFFu + ((u >> 16) & 1u);
  return (unsigned short)(r >> 16);
}

union Frag16 { v16bf bf; u32x4 q[2]; };

// ---------------------------------------------------------------------------
// 1) Transpose x [B,D,T] -> R [N,D] (f32 + bf16), zero Qacc.
// ---------------------------------------------------------------------------
__global__ void rvq_prep(const float* __restrict__ x, float* __restrict__ R,
                         unsigned short* __restrict__ Rb, float* __restrict__ Qacc) {
  __shared__ float tile[32][33];
  int b  = blockIdx.z;
  int d0 = blockIdx.y * 32;
  int t0 = blockIdx.x * 32;
  int tx = threadIdx.x, ty = threadIdx.y;
  #pragma unroll
  for (int i = ty; i < 32; i += 8)
    tile[i][tx] = x[((size_t)b * DD + d0 + i) * TT + t0 + tx];
  __syncthreads();
  #pragma unroll
  for (int i = ty; i < 32; i += 8) {
    size_t o = ((size_t)b * TT + t0 + i) * DD + d0 + tx;
    float v = tile[tx][i];
    R[o] = v;
    Rb[o] = f2bf(v);
    Qacc[o] = 0.0f;
  }
}

// ---------------------------------------------------------------------------
// 2) bf16 codebooks + fp32 ||c||^2 per bin. One block per (q,bin) row.
// ---------------------------------------------------------------------------
__global__ void rvq_cbprep(const float* __restrict__ cb, unsigned short* __restrict__ Cb,
                           float* __restrict__ cnorm) {
  int rb = blockIdx.x;           // q*NB + bin
  int d  = threadIdx.x;          // 0..255
  float c = cb[(size_t)rb * DD + d];
  Cb[(size_t)rb * DD + d] = f2bf(c);
  __shared__ float s[256];
  s[d] = c * c;
  __syncthreads();
  for (int st = 128; st > 0; st >>= 1) {
    if (d < st) s[d] += s[d + st];
    __syncthreads();
  }
  if (d == 0) cnorm[rb] = s[0];
}

// ---------------------------------------------------------------------------
// 3) Fused GEMM + argmin. Wave handles 16 rows vs all 1024 bins.
//    Rank by ||c||^2 - 2*(r.c); ||r||^2 is row-constant so it cannot change
//    the argmin. Codebook chunks are staged into LDS once per workgroup via
//    GLOBAL_LOAD_ASYNC_TO_LDS_B128 (ASYNCcnt), double-buffered.
// ---------------------------------------------------------------------------
__global__ void __launch_bounds__(256) rvq_argmin(
    const unsigned short* __restrict__ Rb, const unsigned short* __restrict__ Cb,
    const float* __restrict__ cnorm, int* __restrict__ idx_out, int q) {
  __shared__ unsigned short sB[2][16 * DD];   // 2 x 8 KB double buffer

  int tid   = threadIdx.x;
  int lane  = tid & 31;
  int wave  = tid >> 5;
  int khalf = lane >> 4;         // 0 or 1
  int lmod  = lane & 15;
  int row0  = blockIdx.x * 128 + wave * 16;  // this wave's first row
  int m     = row0 + lmod;

  // Preload A fragments for all 8 K-steps (16x32 bf16 per step).
  // ISA layout: lane(l) holds M=l%16; K = khalf*8+[0..7] and 16+khalf*8+[0..7].
  Frag16 afrag[8];
  #pragma unroll
  for (int s = 0; s < 8; ++s) {
    const u32x4* p = (const u32x4*)(Rb + (size_t)m * DD + 32 * s + khalf * 8);
    afrag[s].q[0] = p[0];   // k0 + khalf*8     .. +7
    afrag[s].q[1] = p[2];   // k0 + 16+khalf*8  .. +7
  }

  const unsigned char* cbl_bytes = (const unsigned char*)(Cb + (size_t)q * NB * DD);
  const float* cnl = cnorm + q * NB;

  // Async-copy one 16-bin chunk (8192 B) into LDS buffer `buf`:
  // each of the 256 threads moves 32 B with two async b128 transfers.
  // INST_OFFSET is added to BOTH the LDS and global addresses (ISA 10.7/15.18).
  auto prefetch = [&](int buf, int c0) {
    unsigned long long src =
        (unsigned long long)(cbl_bytes + (size_t)c0 * DD * 2 + (size_t)tid * 32);
    unsigned int dst = (unsigned int)(size_t)(const void*)(&sB[buf][0]) + tid * 32;
    asm volatile("global_load_async_to_lds_b128 %0, %1, off"
                 :: "v"(dst), "v"(src) : "memory");
    asm volatile("global_load_async_to_lds_b128 %0, %1, off offset:16"
                 :: "v"(dst), "v"(src) : "memory");
  };

  float minv[8];
  int   mini[8];
  #pragma unroll
  for (int j = 0; j < 8; ++j) { minv[j] = 3.4e38f; mini[j] = 0; }

  prefetch(0, 0);
  asm volatile("s_wait_asynccnt 0" ::: "memory");
  __syncthreads();

  for (int c0 = 0; c0 < NB; c0 += 16) {
    int buf = (c0 >> 4) & 1;
    if (c0 + 16 < NB) prefetch(buf ^ 1, c0 + 16);  // stream next chunk in

    int n = c0 + lmod;                             // this lane's column (bin)
    // B layout: lane(l) holds N=l%16; K = khalf*8+[0..7] and 16+khalf*8+[0..7].
    const unsigned short* brow = &sB[buf][lmod * DD + khalf * 8];
    v8f acc = {};
    #pragma unroll
    for (int s = 0; s < 8; ++s) {
      Frag16 bfrag;
      const u32x4* p = (const u32x4*)(brow + 32 * s);
      bfrag.q[0] = p[0];
      bfrag.q[1] = p[2];
      acc = __builtin_amdgcn_wmma_f32_16x16x32_bf16(
          /*neg_a=*/false, afrag[s].bf, /*neg_b=*/false, bfrag.bf,
          /*c_mod=*/(short)0, acc, /*reuse_a=*/false, /*reuse_b=*/false);
    }
    float cn = cnl[n];
    // C layout: VGPR j holds row (8*khalf + j), column (lane%16).
    #pragma unroll
    for (int j = 0; j < 8; ++j) {
      float dist = cn - 2.0f * acc[j];
      if (dist < minv[j]) { minv[j] = dist; mini[j] = n; }
    }

    // My next-chunk copies done; barrier publishes them to all waves and
    // retires everyone's reads of the buffer we overwrite next iteration.
    asm volatile("s_wait_asynccnt 0" ::: "memory");
    __syncthreads();
  }

  // Argmin reduce across the 16 columns in each lane-half (masks < 16 stay in-half).
  #pragma unroll
  for (int j = 0; j < 8; ++j) {
    #pragma unroll
    for (int off = 1; off < 16; off <<= 1) {
      float ov = __shfl_xor(minv[j], off, 32);
      int   oi = __shfl_xor(mini[j], off, 32);
      if (ov < minv[j] || (ov == minv[j] && oi < mini[j])) { minv[j] = ov; mini[j] = oi; }
    }
    if (lmod == 0) idx_out[row0 + khalf * 8 + j] = mini[j];
  }
}

// ---------------------------------------------------------------------------
// 4) Per-layer update: q = cb[idx]; Qacc += q; loss += (q-r)^2; r -= q.
//    One block per row n; deterministic per-row loss partial (no atomics).
// ---------------------------------------------------------------------------
__global__ void __launch_bounds__(256) rvq_update(
    const float* __restrict__ cb, const int* __restrict__ idx,
    float* __restrict__ R, unsigned short* __restrict__ Rb,
    float* __restrict__ Qacc, float* __restrict__ lossbuf,
    float* __restrict__ codes_out, int q) {
  int n = blockIdx.x;
  int d = threadIdx.x;
  int bi = idx[n];
  float c = cb[((size_t)q * NB + bi) * DD + d];
  size_t o = (size_t)n * DD + d;
  float r = R[o];
  float diff = c - r;
  Qacc[o] += c;
  float rn = r - c;
  R[o] = rn;
  Rb[o] = f2bf(rn);

  __shared__ float s[256];
  s[d] = diff * diff;
  __syncthreads();
  for (int st = 128; st > 0; st >>= 1) {
    if (d < st) s[d] += s[d + st];
    __syncthreads();
  }
  if (d == 0) {
    lossbuf[(size_t)q * NN + n] = s[0];
    codes_out[(size_t)q * NN + n] = (float)bi;
  }
}

// ---------------------------------------------------------------------------
// 5) Transpose Qacc [N,D] -> quantized [B,D,T]; then scalar loss.
// ---------------------------------------------------------------------------
__global__ void rvq_final_q(const float* __restrict__ Qacc, float* __restrict__ outq) {
  __shared__ float tile[32][33];
  int b  = blockIdx.z;
  int d0 = blockIdx.y * 32;
  int t0 = blockIdx.x * 32;
  int tx = threadIdx.x, ty = threadIdx.y;
  #pragma unroll
  for (int i = ty; i < 32; i += 8)
    tile[i][tx] = Qacc[((size_t)b * TT + t0 + i) * DD + d0 + tx];
  __syncthreads();
  #pragma unroll
  for (int i = ty; i < 32; i += 8)
    outq[((size_t)b * DD + d0 + i) * TT + t0 + tx] = tile[tx][i];
}

__global__ void rvq_loss(const float* __restrict__ lossbuf, float* __restrict__ out_loss) {
  __shared__ float s[256];
  int d = threadIdx.x;
  float acc = 0.0f;
  for (size_t i = d; i < (size_t)BQ * NN; i += 256) acc += lossbuf[i];
  s[d] = acc;
  __syncthreads();
  for (int st = 128; st > 0; st >>= 1) {
    if (d < st) s[d] += s[d + st];
    __syncthreads();
  }
  if (d == 0) out_loss[0] = s[0] / ((float)BQ * (float)NN * (float)DD);
}

// ---------------------------------------------------------------------------
// Launch
// ---------------------------------------------------------------------------
extern "C" void kernel_launch(void* const* d_in, const int* in_sizes, int n_in,
                              void* d_out, int out_size, void* d_ws, size_t ws_size,
                              hipStream_t stream) {
  const float* x   = (const float*)d_in[0];   // [B, D, T]
  const float* cbs = (const float*)d_in[1];   // [n_q, bins, D]

  float* out       = (float*)d_out;
  float* out_codes = out + (size_t)BB * DD * TT;   // quantized | codes | loss
  float* out_loss  = out_codes + (size_t)BQ * NN;

  char* w = (char*)d_ws;
  auto take = [&](size_t bytes) { char* p = w; w += (bytes + 255) & ~(size_t)255; return p; };
  float*          R       = (float*)take((size_t)NN * DD * 4);          // 64 MB
  unsigned short* Rb      = (unsigned short*)take((size_t)NN * DD * 2); // 32 MB
  float*          Qacc    = (float*)take((size_t)NN * DD * 4);          // 64 MB
  unsigned short* Cb      = (unsigned short*)take((size_t)BQ * NB * DD * 2); // 4 MB
  float*          cnorm   = (float*)take((size_t)BQ * NB * 4);
  int*            idx     = (int*)take((size_t)NN * 4);
  float*          lossbuf = (float*)take((size_t)BQ * NN * 4);          // 2 MB

  dim3 tb(32, 8, 1);
  dim3 tg(TT / 32, DD / 32, BB);

  rvq_prep<<<tg, tb, 0, stream>>>(x, R, Rb, Qacc);
  rvq_cbprep<<<BQ * NB, 256, 0, stream>>>(cbs, Cb, cnorm);

  for (int q = 0; q < BQ; ++q) {
    rvq_argmin<<<NN / 128, 256, 0, stream>>>(Rb, Cb, cnorm, idx, q);
    rvq_update<<<NN, 256, 0, stream>>>(cbs, idx, R, Rb, Qacc, lossbuf, out_codes, q);
  }

  rvq_final_q<<<tg, tb, 0, stream>>>(Qacc, out);
  rvq_loss<<<1, 256, 0, stream>>>(lossbuf, out_loss);
}
